// QTAttGuided_21620865368155
// MI455X (gfx1250) — matrix operand order repaired and checked by reading, not compiled, not run on hardware
//
#include <hip/hip_runtime.h>

// QTAttGuided for MI455X (gfx1250, wave32).
// Pipeline: 4x transpose(f32 [b,c,M] -> f16 [b,M,c]) -> level0 (coarse, emits
// top-16 -> tp1) -> level1 (fine) -> combine (weight softmax + index perms).
// Workspace layout (bytes), total 48 MiB:
//   kTf  @ 0        : 8 MiB   f16 [2][16384][128]
//   vTf  @ 8 MiB    : 8 MiB
//   kTc  @ 16 MiB   : 2 MiB   f16 [2][4096][128]
//   vTc  @ 18 MiB   : 2 MiB
//   msg0 @ 20 MiB   : 4 MiB   f32 [2][1024][4][8][16]
//   msg1 @ 24 MiB   : 16 MiB  f32 [2][4096][4][8][16]
//   tp1  @ 40 MiB   : 8 MiB   i32 [2][2][4096][16][8]

typedef __attribute__((ext_vector_type(16))) _Float16 v16h;
typedef __attribute__((ext_vector_type(8)))  _Float16 v8h;
typedef __attribute__((ext_vector_type(8)))  float    v8f;

// ---------------------------------------------------------------- transpose
// src: [2][128][M] f32 (channel-major)  ->  dst: [2][M][128] f16 (row = one
// spatial position's 128 channels = 8 heads x 16 dims, contiguous).
__global__ __launch_bounds__(256) void qt_transpose_f16(
    const float* __restrict__ src, _Float16* __restrict__ dst, int M) {
  __shared__ float tile[32][33];
  const int b   = blockIdx.z;
  const int ch0 = blockIdx.y * 32;
  const int p0  = blockIdx.x * 32;
  const int tx = threadIdx.x, ty = threadIdx.y;
  #pragma unroll
  for (int i = 0; i < 32; i += 8)
    tile[ty + i][tx] = src[((size_t)(b * 128 + ch0 + ty + i)) * M + p0 + tx];
  __syncthreads();
  #pragma unroll
  for (int i = 0; i < 32; i += 8)
    dst[((size_t)b * M + p0 + ty + i) * 128 + ch0 + tx] = (_Float16)tile[tx][ty + i];
}

// ---------------------------------------------------------------- level core
// GW: query-group grid width (32 coarse / 64 fine); W1: key grid width
// (64 coarse / 128 fine); TOPK: emit top-16 indices for next level.
// Block = one (b, group); wave w = head h. 8 waves x 32 lanes.
template <int GW, int W1, bool TOPK>
__global__ __launch_bounds__(256) void qt_level(
    const float* __restrict__ qsrc,      // [2][128][2GW][2GW] f32
    const _Float16* __restrict__ kT,     // [2][W1*W1][128] f16
    const _Float16* __restrict__ vT,     // [2][W1*W1][128] f16
    const int* __restrict__ tp,          // [2][2][GW*GW][16][8] i32
    float* __restrict__ msg,             // [2][GW*GW][4][8][16] f32
    int* __restrict__ tpo)               // [2][2][4*GW*GW][16][8] i32 or null
{
  constexpr int NGRP = GW * GW;
  constexpr int W0 = 2 * GW;
  constexpr int M1 = W1 * W1;

  const int bid = blockIdx.x;
  const int b = bid / NGRP;
  const int n = bid % NGRP;
  const int gh = n / GW, gw = n % GW;
  const int tid = threadIdx.x;
  const int h = tid >> 5;            // head = wave id
  const int lane = tid & 31;
  const int lo = lane & 15;
  const bool hi = lane >= 16;

  __shared__ int      idx_s[TOPK ? 8 * 64 : 8];  // flat key pos (top-k only)
  __shared__ float    S_s[8 * 4 * 64];           // raw scaled scores [h][t][cand]
  __shared__ _Float16 P_s[8 * 4 * 64];           // softmax probs (f16 for WMMA)
  __shared__ _Float16 V_s[8 * 64 * 16];          // gathered V rows  [h][cand][d]

  const int* tpr = tp + ((size_t)b * NGRP + n) * 128;   // coord 0 (row picks)
  const int* tpc = tpr + (size_t)2 * NGRP * 128;        // coord 1 (col picks)
  const _Float16* kTb = kT + (size_t)b * M1 * 128;
  const _Float16* vTb = vT + (size_t)b * M1 * 128;

  // --- Phase 1a: this lane's 4 K-tile candidate positions, fully in-register
  // (candidate m = kt*16+lo -> pick kp = kt*4 + lo/4, child (x,y) from lo).
  // No LDS round trip, so all 4 b128 K loads below can be in flight at once.
  int kpos[4];
  {
    const int x = (lo >> 1) & 1, y = lo & 1, kq = lo >> 2;
    #pragma unroll
    for (int kt = 0; kt < 4; ++kt) {
      const int kp = kt * 4 + kq;
      kpos[kt] = (tpr[kp * 8 + h] * 2 + x) * W1 + (tpc[kp * 8 + h] * 2 + y);
    }
  }
  v8h kv[4];
  #pragma unroll
  for (int kt = 0; kt < 4; ++kt)
    kv[kt] = *reinterpret_cast<const v8h*>(
        kTb + (size_t)kpos[kt] * 128 + h * 16 + (hi ? 8 : 0));  // d 0..7 / 8..15

  // --- Phase 1b: gather V rows into LDS with 16B vector loads (2 rows/lane),
  // positions again computed in-register; record flat idx for top-k retrieval.
  #pragma unroll
  for (int e = 0; e < 2; ++e) {
    const int cnd = lane * 2 + e;
    const int kp = cnd >> 2;
    const int x = (cnd >> 1) & 1, y = cnd & 1;
    const int pos = (tpr[kp * 8 + h] * 2 + x) * W1 + (tpc[kp * 8 + h] * 2 + y);
    if (TOPK) idx_s[h * 64 + cnd] = pos;
    const v8h* src = reinterpret_cast<const v8h*>(vTb + (size_t)pos * 128 + h * 16);
    v8h* dst = reinterpret_cast<v8h*>(&V_s[(h * 64 + cnd) * 16]);
    dst[0] = src[0];
    dst[1] = src[1];
  }

  // --- Phase 1c: Q^T as WMMA B operand: N=t<4 lanes hold q[t][d=8h+0..7]
  // in halves 0..7 (K=8h+0..7); halves 8..15 carry K=16..31 which is zero pad.
  v16h qf;
  #pragma unroll
  for (int i = 0; i < 16; ++i) qf[i] = (_Float16)0.f;
  if (lo < 4) {
    const int t1 = lo >> 1, t2 = lo & 1;
    const int row = gh * 2 + t1, col = gw * 2 + t2;
    const float* qp = qsrc + (((size_t)(b * 128 + h * 16)) * W0 + row) * W0 + col;
    #pragma unroll
    for (int dd = 0; dd < 8; ++dd)
      qf[dd] = (_Float16)qp[(size_t)(dd + (hi ? 8 : 0)) * W0 * W0];
  }

  // --- Phase 2: scores. A = gathered K tile (M=16 candidates, K=d zero-
  // padded to 32); loads were issued above, waits stagger loadcnt 3..0.
  #pragma unroll
  for (int kt = 0; kt < 4; ++kt) {
    v16h af;
    #pragma unroll
    for (int i = 0; i < 8; ++i) af[i] = kv[kt][i];
    #pragma unroll
    for (int i = 8; i < 16; ++i) af[i] = (_Float16)0.f;   // K=16..31 pad
    v8f c;
    #pragma unroll
    for (int r = 0; r < 8; ++r) c[r] = 0.f;
    c = __builtin_amdgcn_wmma_f32_16x16x32_f16(false, af, false, qf,
                                               (short)0, c, false, false);
    if (lo < 4) {  // N=t column valid; VGPR r -> cand kt*16 + r (+8 hi half)
      const int cb = kt * 16 + (hi ? 8 : 0);
      #pragma unroll
      for (int r = 0; r < 8; ++r)
        S_s[(h * 4 + lo) * 64 + cb + r] = c[r] * 0.25f;    // d^-0.5
    }
  }
  __syncthreads();

  // --- Phase 3: softmax over 64 candidates; 8-lane group per t
  const int g = lane >> 3, j = lane & 7;
  const int sb = (h * 4 + g) * 64;
  float vals[8];
  #pragma unroll
  for (int i = 0; i < 8; ++i) vals[i] = S_s[sb + i * 8 + j];
  float mx = vals[0];
  #pragma unroll
  for (int i = 1; i < 8; ++i) mx = fmaxf(mx, vals[i]);
  #pragma unroll
  for (int off = 1; off < 8; off <<= 1) mx = fmaxf(mx, __shfl_xor(mx, off, 32));
  float ev[8], sum = 0.f;
  #pragma unroll
  for (int i = 0; i < 8; ++i) { ev[i] = __expf(vals[i] - mx); sum += ev[i]; }
  #pragma unroll
  for (int off = 1; off < 8; off <<= 1) sum += __shfl_xor(sum, off, 32);
  const float inv = 1.f / sum;
  #pragma unroll
  for (int i = 0; i < 8; ++i) P_s[sb + i * 8 + j] = (_Float16)(ev[i] * inv);

  if (TOPK) {
    // Top-16 on raw scores (order-equivalent to softmax probs). Selection
    // order is irrelevant downstream (level-1 sums over candidates).
    const int p = (gh * 2 + (g >> 1)) * W0 + (gw * 2 + (g & 1));
    constexpr int NQ = 4 * NGRP;
    #pragma unroll 1
    for (int r = 0; r < 16; ++r) {
      float bv = -3.4e38f; int bc = 64;
      #pragma unroll
      for (int i = 0; i < 8; ++i) {
        const int c = i * 8 + j;
        if (vals[i] > bv || (vals[i] == bv && c < bc)) { bv = vals[i]; bc = c; }
      }
      #pragma unroll
      for (int off = 1; off < 8; off <<= 1) {
        const float ov = __shfl_xor(bv, off, 32);
        const int   oc = __shfl_xor(bc, off, 32);
        if (ov > bv || (ov == bv && oc < bc)) { bv = ov; bc = oc; }
      }
      if (j == 0) {
        const int flat = idx_s[h * 64 + bc];
        tpo[((size_t)b * NQ + p) * 128 + r * 8 + h] = flat / W1;
        tpo[(size_t)2 * NQ * 128 + ((size_t)b * NQ + p) * 128 + r * 8 + h] = flat % W1;
      }
      if ((bc & 7) == j) vals[bc >> 3] = -3.4e38f;   // owner retires winner
    }
  }
  __syncthreads();

  // --- Phase 4: message = P . V. A = V^T (M=d=16, K=cand), B = P^T (N=t).
  v8f acc;
  #pragma unroll
  for (int r = 0; r < 8; ++r) acc[r] = 0.f;
  #pragma unroll
  for (int cbk = 0; cbk < 2; ++cbk) {         // K blocks of 32 candidates
    v16h af;
    #pragma unroll
    for (int hb = 0; hb < 2; ++hb) {
      const int cbase = cbk * 32 + (hi ? 16 : 0) + hb * 8;
      #pragma unroll
      for (int i = 0; i < 8; ++i)
        af[hb * 8 + i] = V_s[(h * 64 + cbase + i) * 16 + lo];
    }
    v16h bf;
    if (lo < 4) {
      const int base = (h * 4 + lo) * 64 + cbk * 32 + (hi ? 16 : 0);
      #pragma unroll
      for (int s = 0; s < 16; ++s) bf[s] = P_s[base + s];
    } else {
      #pragma unroll
      for (int s = 0; s < 16; ++s) bf[s] = (_Float16)0.f;
    }
    acc = __builtin_amdgcn_wmma_f32_16x16x32_f16(false, af, false, bf,
                                                 (short)0, acc, false, false);
  }
  if (lo < 4) {   // N=t column; VGPR r -> dd = r (+8 for hi half)
    float* mo = msg + ((((size_t)b * NGRP + n) * 4 + lo) * 8 + h) * 16 + (hi ? 8 : 0);
    #pragma unroll
    for (int r = 0; r < 8; ++r) mo[r] = acc[r];
  }
}

// ---------------------------------------------------------------- combine
// out[b, fp, h, dd] = w0*msg0[b, n0(p), tC(p), h, dd] + w1*msg1[b, p, tF, h, dd]
// with fp = ((i64a*2+t1)*64 + i64b)*2 + t2 (comb H=64) and the faithful
// H=128 comb quirk for msg0: p split as ((i128*2+t1c)*8 + i8)*2 + t2c.
__global__ __launch_bounds__(256) void qt_combine(
    const float* __restrict__ msg0, const float* __restrict__ msg1,
    const float* __restrict__ weight, float* __restrict__ out) {
  const int gid = blockIdx.x * 256 + threadIdx.x;   // 2*16384*128 elements
  const int c  = gid & 127;
  const int rest = gid >> 7;
  const int fp = rest & 16383;
  const int b  = rest >> 14;

  const float w0v = weight[0], w1v = weight[1];
  const float mw = fmaxf(w0v, w1v);
  const float e0 = __expf(w0v - mw), e1 = __expf(w1v - mw);
  const float s = 1.f / (e0 + e1);
  const float w0 = e0 * s, w1 = e1 * s;

  const int t2 = fp & 1;
  int u = fp >> 1;
  const int i64b = u & 63; u >>= 6;
  const int t1 = u & 1;
  const int i64a = u >> 1;
  const int p  = i64a * 64 + i64b;
  const int tF = t1 * 2 + t2;

  const int t2c = p & 1;
  int v = p >> 1;
  const int i8 = v & 7; v >>= 3;
  const int t1c = v & 1;
  const int i128 = v >> 1;
  const int n0 = i128 * 8 + i8;
  const int tC = t1c * 2 + t2c;

  const float m0 = msg0[(((size_t)b * 1024 + n0) * 4 + tC) * 128 + c];
  const float m1 = msg1[(((size_t)b * 4096 + p) * 4 + tF) * 128 + c];
  out[gid] = w0 * m0 + w1 * m1;
}

// ---------------------------------------------------------------- launcher
extern "C" void kernel_launch(void* const* d_in, const int* in_sizes, int n_in,
                              void* d_out, int out_size, void* d_ws, size_t ws_size,
                              hipStream_t stream) {
  const float* q_fine   = (const float*)d_in[0];
  const float* q_coarse = (const float*)d_in[1];
  const float* k_fine   = (const float*)d_in[2];
  const float* k_coarse = (const float*)d_in[3];
  const float* v_fine   = (const float*)d_in[4];
  const float* v_coarse = (const float*)d_in[5];
  const float* weight   = (const float*)d_in[6];
  const int*   topk_pos = (const int*)d_in[7];

  char* ws = (char*)d_ws;
  const size_t MB = 1024u * 1024u;
  _Float16* kTf = (_Float16*)(ws + 0);
  _Float16* vTf = (_Float16*)(ws + 8 * MB);
  _Float16* kTc = (_Float16*)(ws + 16 * MB);
  _Float16* vTc = (_Float16*)(ws + 18 * MB);
  float*    msg0 = (float*)(ws + 20 * MB);
  float*    msg1 = (float*)(ws + 24 * MB);
  int*      tp1  = (int*)(ws + 40 * MB);

  dim3 tb(32, 8);
  qt_transpose_f16<<<dim3(16384 / 32, 4, 2), tb, 0, stream>>>(k_fine, kTf, 16384);
  qt_transpose_f16<<<dim3(16384 / 32, 4, 2), tb, 0, stream>>>(v_fine, vTf, 16384);
  qt_transpose_f16<<<dim3(4096 / 32, 4, 2),  tb, 0, stream>>>(k_coarse, kTc, 4096);
  qt_transpose_f16<<<dim3(4096 / 32, 4, 2),  tb, 0, stream>>>(v_coarse, vTc, 4096);

  // Level 0: coarse (GW=32 groups, 64x64 key grid), emits top-16 -> tp1
  qt_level<32, 64, true><<<2 * 1024, 256, 0, stream>>>(
      q_coarse, kTc, vTc, topk_pos, msg0, tp1);
  // Level 1: fine (GW=64 groups, 128x128 key grid)
  qt_level<64, 128, false><<<2 * 4096, 256, 0, stream>>>(
      q_fine, kTf, vTf, tp1, msg1, nullptr);

  qt_combine<<<(2 * 16384 * 128) / 256, 256, 0, stream>>>(msg0, msg1, weight,
                                                          (float*)d_out);
}
